// TimeLSTM_56530359550494
// MI455X (gfx1250) — matrix operand order repaired and checked by reading, not compiled, not run on hardware
//
#include <hip/hip_runtime.h>
#include <hip/hip_bf16.h>
#include <math.h>

// ---------------------------------------------------------------------------
// Time-LSTM fused recurrence for MI455X (gfx1250, wave32, WMMA bf16)
//   B=1024 batch rows, L=200 steps, D=128, gates=4D=512
//   grid = 32 WGs x 256 threads (8 waves); each WG owns 32 batch rows,
//   iterates L steps with v_wmma_f32_16x16x32_bf16 GEMMs out of LDS.
//   All transcendentals are branchless native v_exp_f32 / v_rcp_f32.
// ---------------------------------------------------------------------------

typedef __attribute__((ext_vector_type(16))) __bf16 v16bf;
typedef __attribute__((ext_vector_type(8)))  float  v8f;

#define WMMA_BF16(a, b, c) \
  __builtin_amdgcn_wmma_f32_16x16x32_bf16(false, (a), false, (b), (short)0, (c), false, false)

constexpr int kB = 1024;
constexpr int kL = 200;
constexpr int kD = 128;
constexpr int kG = 512;     // 4*D gate width
constexpr int kTile = 32;   // batch rows per workgroup
constexpr int kThreads = 256;

// Branchless sigmoid: rcp(1 + exp(-x)).  exp(-x)->inf for very negative x,
// rcp(inf)=0, so the limit behavior is correct without range branches.
__device__ __forceinline__ float fast_sigmoid(float x) {
  return __builtin_amdgcn_rcpf(1.0f + __expf(-x));
}

// Branchless tanh: e = exp(-2|x|) in (0,1], r = (1-e)/(1+e), restore sign.
__device__ __forceinline__ float fast_tanh(float x) {
  float e = __expf(-2.0f * __builtin_fabsf(x));
  float r = (1.0f - e) * __builtin_amdgcn_rcpf(1.0f + e);
  return __builtin_copysignf(r, x);
}

// A fragment (16x32 bf16, MxK). Arow = &A[m][0] (m = lane&15), half = lane>>4.
// Layout: lanes 0-15 hold K kbase..+15, lanes 16-31 hold K kbase+16..+31.
__device__ __forceinline__ v16bf load_a_frag(const __bf16* Arow, int kbase, int half) {
  const __bf16* p = Arow + kbase + half * 16;
  v16bf a;
#pragma unroll
  for (int e = 0; e < 16; ++e) a[e] = p[e];
  return a;
}

// B fragment (32x16 bf16, KxN) from row-major weight W[n][k] (out = in @ W^T).
// Wrow = &W[nblk*16 + (lane&15)][0]; lanes 0-15: K kbase..+15, lanes 16-31: +16.
__device__ __forceinline__ v16bf load_b_frag(const __bf16* Wrow, int kbase, int half) {
  const __bf16* p = Wrow + kbase + half * 16;
  v16bf b;
#pragma unroll
  for (int e = 0; e < 16; ++e) b[e] = p[e];
  return b;
}

__global__ __launch_bounds__(kThreads)
void timelstm_fused_kernel(const int* __restrict__ log_seqs,
                           const int* __restrict__ pos_seqs,
                           const int* __restrict__ neg_seqs,
                           const float* __restrict__ item_emb,
                           const float* __restrict__ W_all_w,
                           const float* __restrict__ W_all_b,
                           const float* __restrict__ U_all_w,
                           const float* __restrict__ U_all_b,
                           const float* __restrict__ W_d_w,
                           const float* __restrict__ W_d_b,
                           float* __restrict__ out) {
  // ---- LDS (~315.5 KB of the 320 KB WGP budget) ----
  __shared__ __bf16 sWcat[kG][256];    // [n][k] : k<128 -> W_all, k>=128 -> U_all (256 KB)
  __shared__ __bf16 sWd[kD][kD];       // W_d row-major                           (32 KB)
  __shared__ __bf16 sA[kTile][256];    // A operand: [h(0:128) | emb(128:256)]    (16 KB)
  __shared__ __bf16 sCb[kTile][kD];    // bf16 copy of c state                    (8 KB)
  __shared__ float  sBias[kG];         // W_all_b + U_all_b
  __shared__ float  sWdB[kD];
  __shared__ float  sT[kTile];
  __shared__ int    sPid[kTile];
  __shared__ int    sNid[kTile];
  __shared__ float  sLogit[2][kTile];

  const int tid  = threadIdx.x;
  const int lane = tid & 31;
  const int w    = tid >> 5;     // wave id 0..7 -> owns d-slice [16w, 16w+16)
  const int col  = lane & 15;    // N column within 16-wide block / A row in M-block
  const int hsel = lane >> 4;    // lane half
  const int wgBase = blockIdx.x * kTile;

  // ---- one-time weight conversion fp32 -> bf16 into LDS ----
  for (int i = tid; i < kG * 256; i += kThreads) {
    int n = i >> 8, k = i & 255;
    float v = (k < kD) ? W_all_w[n * kD + k] : U_all_w[n * kD + (k - kD)];
    sWcat[n][k] = (__bf16)v;
  }
  for (int i = tid; i < kD * kD; i += kThreads)
    sWd[i >> 7][i & 127] = (__bf16)W_d_w[i];
  for (int i = tid; i < kG; i += kThreads)
    sBias[i] = W_all_b[i] + U_all_b[i];
  if (tid < kD) sWdB[tid] = W_d_b[tid];
  for (int i = tid; i < kTile * kD; i += kThreads) {
    sA[i >> 7][i & 127] = (__bf16)0.0f;   // h0 = 0
    sCb[i >> 7][i & 127] = (__bf16)0.0f;  // c0 = 0
  }
  __syncthreads();

  // ---- loop-invariant register state ----
  // W_d B-fragments for this wave's d-slice (4 x v16bf = 32 VGPRs), biases.
  v16bf wdfrag[4];
#pragma unroll
  for (int ks = 0; ks < 4; ++ks)
    wdfrag[ks] = load_b_frag(&sWd[w * 16 + col][0], ks * 32, hsel);
  float biasReg[4];
#pragma unroll
  for (int g = 0; g < 4; ++g)
    biasReg[g] = sBias[(g * 8 + w) * 16 + col];
  const float wdbReg = sWdB[w * 16 + col];

  v8f cfrag[2];
#pragma unroll
  for (int mb = 0; mb < 2; ++mb)
#pragma unroll
    for (int v = 0; v < 8; ++v) cfrag[mb][v] = 0.0f;

  for (int l = 0; l < kL; ++l) {
    __syncthreads();  // previous step fully consumed (incl. sLogit output reads)

    // ---- stage: gather this step's embedding rows (bf16), ts, pos/neg ids ----
    {
      int r = tid >> 3, q = tid & 7;           // 8 threads per batch row
      long base = ((long)(wgBase + r) * kL + l) * 2;
      int id = log_seqs[base];
      const float* ep = item_emb + (long)id * kD + q * 16;
#pragma unroll
      for (int j = 0; j < 16; ++j)
        sA[r][kD + q * 16 + j] = (__bf16)ep[j];
      if (l + 1 < kL) {  // prefetch next step's embedding row (global_prefetch_b8)
        int idn = log_seqs[base + 2];
        __builtin_prefetch(item_emb + (long)idn * kD + q * 16, 0, 0);
      }
      if (q == 0) {
        sT[r]   = (float)log_seqs[base + 1];
        sPid[r] = pos_seqs[base];
        sNid[r] = neg_seqs[base];
      }
      if (tid < 2 * kTile) ((float*)sLogit)[tid] = 0.0f;
    }
    __syncthreads();

    // ---- c_s1 = tanh(c @ W_d^T + b);  c_adj = c - c_s1 + c_s1 * t ----
    v8f cadj[2];
#pragma unroll
    for (int mb = 0; mb < 2; ++mb) {
      v8f cs = {wdbReg, wdbReg, wdbReg, wdbReg, wdbReg, wdbReg, wdbReg, wdbReg};
#pragma unroll
      for (int ks = 0; ks < 4; ++ks) {
        v16bf a = load_a_frag(&sCb[mb * 16 + col][0], ks * 32, hsel);
        cs = WMMA_BF16(a, wdfrag[ks], cs);
      }
#pragma unroll
      for (int v = 0; v < 8; ++v) {
        float s1 = fast_tanh(cs[v]);
        float tv = sT[mb * 16 + v + hsel * 8];
        cadj[mb][v] = cfrag[mb][v] - s1 + s1 * tv;
      }
    }

    // ---- gates: [h|emb](32x256) @ [W_all|U_all]^T ----
    // wave w owns gate N-blocks {w, w+8, w+16, w+24} so f/i/o/c~ for the same
    // (row, d) sit in the same lane/VGPR slot -> gate math stays in registers.
    v8f ofrag[2];
#pragma unroll
    for (int mb = 0; mb < 2; ++mb) {
      v8f acc[4];
#pragma unroll
      for (int g = 0; g < 4; ++g) {
        float bg = biasReg[g];
        acc[g] = (v8f){bg, bg, bg, bg, bg, bg, bg, bg};
      }
#pragma unroll
      for (int ks = 0; ks < 8; ++ks) {
        v16bf a = load_a_frag(&sA[mb * 16 + col][0], ks * 32, hsel);
#pragma unroll
        for (int g = 0; g < 4; ++g) {
          v16bf bb = load_b_frag(&sWcat[(g * 8 + w) * 16 + col][0], ks * 32, hsel);
          acc[g] = WMMA_BF16(a, bb, acc[g]);
        }
      }
#pragma unroll
      for (int v = 0; v < 8; ++v) {
        float f  = fast_sigmoid(acc[0][v]);
        float i  = fast_sigmoid(acc[1][v]);
        float o  = fast_sigmoid(acc[2][v]);
        float ct = fast_sigmoid(acc[3][v]);
        float cn = f * cadj[mb][v] + i * ct;
        cfrag[mb][v] = cn;
        ofrag[mb][v] = o;
      }
    }

    __syncthreads();  // everyone done reading sA / sCb

    // ---- write back bf16 h_new, c_new for next step's A operands ----
#pragma unroll
    for (int mb = 0; mb < 2; ++mb) {
#pragma unroll
      for (int v = 0; v < 8; ++v) {
        int row = mb * 16 + v + hsel * 8;
        float hn = ofrag[mb][v] * fast_tanh(cfrag[mb][v]);
        sA[row][w * 16 + col]  = (__bf16)hn;
        sCb[row][w * 16 + col] = (__bf16)cfrag[mb][v];
      }
    }

    // ---- on-the-fly logits: o . pos_emb / o . neg_emb (L2-resident gathers) ----
    {
      int d = w * 16 + col;
#pragma unroll
      for (int mb = 0; mb < 2; ++mb) {
#pragma unroll
        for (int v = 0; v < 8; ++v) {
          int row = mb * 16 + v + hsel * 8;
          float o  = ofrag[mb][v];
          float pv = o * item_emb[(long)sPid[row] * kD + d];
          float nv = o * item_emb[(long)sNid[row] * kD + d];
#pragma unroll
          for (int off = 8; off > 0; off >>= 1) {   // reduce 16-lane d-group
            pv += __shfl_xor(pv, off, 16);
            nv += __shfl_xor(nv, off, 16);
          }
          if (col == 0) {
            atomicAdd(&sLogit[0][row], pv);
            atomicAdd(&sLogit[1][row], nv);
          }
        }
      }
    }
    __syncthreads();

    if (tid < kTile) {
      long b = wgBase + tid;
      out[b * kL + l]                 = sLogit[0][tid];  // pos_logits
      out[(long)kB * kL + b * kL + l] = sLogit[1][tid];  // neg_logits
    }
  }
}

extern "C" void kernel_launch(void* const* d_in, const int* in_sizes, int n_in,
                              void* d_out, int out_size, void* d_ws, size_t ws_size,
                              hipStream_t stream) {
  (void)in_sizes; (void)n_in; (void)out_size; (void)d_ws; (void)ws_size;
  const int*   log_seqs = (const int*)d_in[1];
  const int*   pos_seqs = (const int*)d_in[2];
  const int*   neg_seqs = (const int*)d_in[3];
  const float* item_emb = (const float*)d_in[4];
  const float* W_all_w  = (const float*)d_in[5];
  const float* W_all_b  = (const float*)d_in[6];
  const float* U_all_w  = (const float*)d_in[7];
  const float* U_all_b  = (const float*)d_in[8];
  const float* W_d_w    = (const float*)d_in[9];
  const float* W_d_b    = (const float*)d_in[10];
  float* out = (float*)d_out;

  dim3 grid(kB / kTile);   // 32 workgroups
  dim3 block(kThreads);    // 8 wave32 waves
  timelstm_fused_kernel<<<grid, block, 0, stream>>>(
      log_seqs, pos_seqs, neg_seqs, item_emb,
      W_all_w, W_all_b, U_all_w, U_all_b, W_d_w, W_d_b, out);
}
